// GNNMaterialPatchModel_35218731827626
// MI455X (gfx1250) — matrix operand order repaired and checked by reading, not compiled, not run on hardware
//
#include <hip/hip_runtime.h>

// ---------------------------------------------------------------------------
// GNN Encode-Process-Decode on gfx1250 (wave32, WMMA bf16 -> f32)
// ---------------------------------------------------------------------------
constexpr int N_ = 50000;   // nodes
constexpr int E_ = 800000;  // edges   (divisible by 64)
constexpr int H_ = 64;      // hidden
constexpr int L_ = 2;       // message passing steps
constexpr int FN_ = 8, FE_ = 4, FOUT_ = 6;

typedef __attribute__((ext_vector_type(16))) __bf16 v16bf;
typedef __attribute__((ext_vector_type(8)))  float  v8f;

// fp32 -> bf16 round-to-nearest-even
__device__ __forceinline__ __bf16 f2bf(float f) {
    unsigned u = __builtin_bit_cast(unsigned, f);
    unsigned r = u + 0x7fffu + ((u >> 16) & 1u);
    unsigned short h = (unsigned short)(r >> 16);
    return __builtin_bit_cast(__bf16, h);
}

// A-operand fragment, 16x32 bf16, from row-major tile (rows across lanes).
// ISA layout (CDNA5 ISA, 16-bit A 16x32): lane m (0-15) holds
// k = k0..k0+7 and k0+16..k0+23 ; lane m+16 holds k0+8..k0+15 and k0+24..k0+31.
__device__ __forceinline__ v16bf load_frag_a(const __bf16* tile, int ld, int lane) {
    int rc = lane & 15;
    int kh = (lane >> 4) & 1;
    const __bf16* p = tile + rc * ld + kh * 8;
    union { uint4 q[2]; v16bf v; } u;
    u.q[0] = *reinterpret_cast<const uint4*>(p);
    u.q[1] = *reinterpret_cast<const uint4*>(p + 16);
    return u.v;
}

// B-operand fragment, 32x16 bf16 (K x N), loaded from W^T[n][k] row-major.
// ISA layout (CDNA5 ISA, 16-bit B): lane n (0-15) holds
// k = k0..k0+7 and k0+16..k0+23 of column n ; lane n+16 holds k0+8..k0+15 and k0+24..k0+31.
__device__ __forceinline__ v16bf load_frag_b(const __bf16* wT, int ld, int lane) {
    int rc = lane & 15;
    int kh = (lane >> 4) & 1;
    const __bf16* p = wT + rc * ld + kh * 8;
    union { uint4 q[2]; v16bf v; } u;
    u.q[0] = *reinterpret_cast<const uint4*>(p);
    u.q[1] = *reinterpret_cast<const uint4*>(p + 16);
    return u.v;
}

__device__ __forceinline__ v8f wmma_bf16(v16bf a, v16bf b, v8f c) {
    return __builtin_amdgcn_wmma_f32_16x16x32_bf16(
        /*neg_a=*/false, a, /*neg_b=*/false, b,
        /*c_mod=*/(short)0, c, /*reuse_a=*/false, /*reuse_b=*/false);
}

// Layer 1: h = relu(in @ W1 + b1) for this wave's 16 rows, all 64 cols.
// ldsIn: [64][K1] bf16 ; W1T: [64][K1] bf16 (transposed) ; writes ldsH [64][64] bf16.
template<int K1>
__device__ __forceinline__ void wave_mlp_layer1(const __bf16* ldsIn, const __bf16* W1T,
                                                const float* b1, __bf16* ldsH,
                                                int m0, int lane) {
    v8f acc[4] = {};
    #pragma unroll
    for (int k0 = 0; k0 < K1; k0 += 32) {
        v16bf a = load_frag_a(ldsIn + m0 * K1 + k0, K1, lane);
        #pragma unroll
        for (int t = 0; t < 4; ++t) {
            v16bf b = load_frag_b(W1T + (t * 16) * K1 + k0, K1, lane);
            acc[t] = wmma_bf16(a, b, acc[t]);
        }
    }
    // C layout: lane<16 -> n=lane, m=j ; lane>=16 -> n=lane-16, m=j+8
    int n = lane & 15, hi = lane >> 4;
    #pragma unroll
    for (int t = 0; t < 4; ++t) {
        float bias = b1[t * 16 + n];
        #pragma unroll
        for (int j = 0; j < 8; ++j) {
            int m = hi ? j + 8 : j;
            float v = fmaxf(acc[t][j] + bias, 0.f);   // ReLU -> v_max_num_f32
            ldsH[(m0 + m) * 64 + t * 16 + n] = f2bf(v);
        }
    }
}

// Layer 2 GEMM, K = 64: acc += ldsH @ W2 (bias/residual handled by caller)
__device__ __forceinline__ void wave_gemm_k64(const __bf16* ldsH, const __bf16* W2T,
                                              v8f (&acc)[4], int m0, int lane) {
    #pragma unroll
    for (int k0 = 0; k0 < 64; k0 += 32) {
        v16bf a = load_frag_a(ldsH + m0 * 64 + k0, 64, lane);
        #pragma unroll
        for (int t = 0; t < 4; ++t) {
            v16bf b = load_frag_b(W2T + (t * 16) * 64 + k0, 64, lane);
            acc[t] = wmma_bf16(a, b, acc[t]);
        }
    }
}

// ---------------------------------------------------------------------------
// Weight prep: fp32 [K][Nc] -> bf16 transposed, zero-padded [Npad][Kpad]
// ---------------------------------------------------------------------------
__global__ void k_cvt_wT(const float* __restrict__ W, __bf16* __restrict__ Wt,
                         int K, int Nc, int Kpad, int Npad) {
    int idx = blockIdx.x * blockDim.x + threadIdx.x;
    if (idx >= Kpad * Npad) return;
    int n = idx / Kpad, k = idx % Kpad;
    float v = (n < Nc && k < K) ? W[k * Nc + n] : 0.f;
    Wt[idx] = f2bf(v);
}

__global__ void k_zero(float* __restrict__ p, int n) {
    int i = blockIdx.x * blockDim.x + threadIdx.x;
    if (i < n) p[i] = 0.f;
}

// ---------------------------------------------------------------------------
// Node encoder: x[N,8] -> x_h[N,64]   (K1 padded 8->32)
// ---------------------------------------------------------------------------
__global__ void k_encode_nodes(const float* __restrict__ x,
                               const __bf16* __restrict__ W1T, const float* __restrict__ b1,
                               const __bf16* __restrict__ W2T, const float* __restrict__ b2,
                               float* __restrict__ x_h) {
    __shared__ __bf16 ldsIn[64 * 32];
    __shared__ __bf16 ldsH[64 * 64];
    int tid = threadIdx.x;
    int n0 = blockIdx.x * 64;
    int r = tid >> 1, hf = tid & 1;
    int node = n0 + r;
    #pragma unroll
    for (int j = 0; j < 16; ++j) {
        int c = hf * 16 + j;
        float v = (node < N_ && c < FN_) ? x[node * FN_ + c] : 0.f;
        ldsIn[r * 32 + c] = f2bf(v);
    }
    __syncthreads();
    int wv = tid >> 5, lane = tid & 31, m0 = wv * 16;
    wave_mlp_layer1<32>(ldsIn, W1T, b1, ldsH, m0, lane);
    __syncthreads();
    v8f acc[4] = {};
    wave_gemm_k64(ldsH, W2T, acc, m0, lane);
    int n = lane & 15, hi = lane >> 4;
    #pragma unroll
    for (int t = 0; t < 4; ++t) {
        float bias = b2[t * 16 + n];
        #pragma unroll
        for (int j = 0; j < 8; ++j) {
            int m = hi ? j + 8 : j;
            int nd = n0 + m0 + m;
            if (nd < N_) x_h[nd * 64 + t * 16 + n] = acc[t][j] + bias;
        }
    }
}

// ---------------------------------------------------------------------------
// Edge encoder: edge_attr[E,4] -> e_h[E,64]   (K1 padded 4->32, E % 64 == 0)
// ---------------------------------------------------------------------------
__global__ void k_encode_edges(const float* __restrict__ ea,
                               const __bf16* __restrict__ W1T, const float* __restrict__ b1,
                               const __bf16* __restrict__ W2T, const float* __restrict__ b2,
                               float* __restrict__ e_h) {
    __shared__ __bf16 ldsIn[64 * 32];
    __shared__ __bf16 ldsH[64 * 64];
    int tid = threadIdx.x;
    int e0 = blockIdx.x * 64;
    int r = tid >> 1, hf = tid & 1;
    int e = e0 + r;
    #pragma unroll
    for (int j = 0; j < 16; ++j) {
        int c = hf * 16 + j;
        float v = (c < FE_) ? ea[e * FE_ + c] : 0.f;
        ldsIn[r * 32 + c] = f2bf(v);
    }
    __syncthreads();
    int wv = tid >> 5, lane = tid & 31, m0 = wv * 16;
    wave_mlp_layer1<32>(ldsIn, W1T, b1, ldsH, m0, lane);
    __syncthreads();
    v8f acc[4] = {};
    wave_gemm_k64(ldsH, W2T, acc, m0, lane);
    int n = lane & 15, hi = lane >> 4;
    #pragma unroll
    for (int t = 0; t < 4; ++t) {
        float bias = b2[t * 16 + n];
        #pragma unroll
        for (int j = 0; j < 8; ++j) {
            int m = hi ? j + 8 : j;
            e_h[(e0 + m0 + m) * 64 + t * 16 + n] = acc[t][j] + bias;
        }
    }
}

// ---------------------------------------------------------------------------
// Processor edge step (fused): gather concat(e_h, x_h[src], x_h[dst]) ->
// MLP(192->64->64) -> residual e_h -> scatter-add into agg[dst]
// ---------------------------------------------------------------------------
__global__ void k_edge_step(const int* __restrict__ eidx,
                            const float* __restrict__ x_h,
                            float* __restrict__ e_h, float* __restrict__ agg,
                            const __bf16* __restrict__ W1T, const float* __restrict__ b1,
                            const __bf16* __restrict__ W2T, const float* __restrict__ b2) {
    __shared__ __bf16 ldsIn[64 * 192];
    __shared__ __bf16 ldsH[64 * 64];
    int tid = threadIdx.x;
    int e0 = blockIdx.x * 64;
    int r = tid >> 1, hf = tid & 1;
    int e = e0 + r;
    int s = eidx[e], d = eidx[E_ + e];
    #pragma unroll
    for (int j = 0; j < 24; ++j) {
        int c = hf * 96 + j * 4;
        const float* sp = (c < 64)  ? (e_h + e * 64 + c)
                        : (c < 128) ? (x_h + s * 64 + (c - 64))
                                    : (x_h + d * 64 + (c - 128));
        float4 v = *reinterpret_cast<const float4*>(sp);
        ldsIn[r * 192 + c + 0] = f2bf(v.x);
        ldsIn[r * 192 + c + 1] = f2bf(v.y);
        ldsIn[r * 192 + c + 2] = f2bf(v.z);
        ldsIn[r * 192 + c + 3] = f2bf(v.w);
    }
    __syncthreads();
    int wv = tid >> 5, lane = tid & 31, m0 = wv * 16;
    wave_mlp_layer1<192>(ldsIn, W1T, b1, ldsH, m0, lane);
    __syncthreads();
    v8f acc[4] = {};
    wave_gemm_k64(ldsH, W2T, acc, m0, lane);
    int n = lane & 15, hi = lane >> 4;
    int dsts[8];
    #pragma unroll
    for (int j = 0; j < 8; ++j) {
        int m = hi ? j + 8 : j;
        dsts[j] = eidx[E_ + e0 + m0 + m];
    }
    #pragma unroll
    for (int t = 0; t < 4; ++t) {
        float bias = b2[t * 16 + n];
        #pragma unroll
        for (int j = 0; j < 8; ++j) {
            int m = hi ? j + 8 : j;
            int ee = e0 + m0 + m;
            int col = t * 16 + n;
            float v = acc[t][j] + bias + e_h[ee * 64 + col];   // residual
            e_h[ee * 64 + col] = v;
            atomicAdd(&agg[dsts[j] * 64 + col], v);            // segment_sum
        }
    }
}

// ---------------------------------------------------------------------------
// Processor node step: concat(x_h, agg) -> MLP(128->64->64) -> residual x_h
// ---------------------------------------------------------------------------
__global__ void k_node_step(const float* __restrict__ agg, float* __restrict__ x_h,
                            const __bf16* __restrict__ W1T, const float* __restrict__ b1,
                            const __bf16* __restrict__ W2T, const float* __restrict__ b2) {
    __shared__ __bf16 ldsIn[64 * 128];
    __shared__ __bf16 ldsH[64 * 64];
    int tid = threadIdx.x;
    int n0 = blockIdx.x * 64;
    int r = tid >> 1, hf = tid & 1;
    int node = n0 + r;
    #pragma unroll
    for (int j = 0; j < 16; ++j) {
        int c = hf * 64 + j * 4;
        float4 v = make_float4(0.f, 0.f, 0.f, 0.f);
        if (node < N_) {
            const float* sp = (c < 64) ? (x_h + node * 64 + c) : (agg + node * 64 + (c - 64));
            v = *reinterpret_cast<const float4*>(sp);
        }
        ldsIn[r * 128 + c + 0] = f2bf(v.x);
        ldsIn[r * 128 + c + 1] = f2bf(v.y);
        ldsIn[r * 128 + c + 2] = f2bf(v.z);
        ldsIn[r * 128 + c + 3] = f2bf(v.w);
    }
    __syncthreads();
    int wv = tid >> 5, lane = tid & 31, m0 = wv * 16;
    wave_mlp_layer1<128>(ldsIn, W1T, b1, ldsH, m0, lane);
    __syncthreads();
    v8f acc[4] = {};
    wave_gemm_k64(ldsH, W2T, acc, m0, lane);
    int n = lane & 15, hi = lane >> 4;
    #pragma unroll
    for (int t = 0; t < 4; ++t) {
        float bias = b2[t * 16 + n];
        #pragma unroll
        for (int j = 0; j < 8; ++j) {
            int m = hi ? j + 8 : j;
            int nd = n0 + m0 + m;
            if (nd < N_) {
                int col = t * 16 + n;
                x_h[nd * 64 + col] = acc[t][j] + bias + x_h[nd * 64 + col];  // residual
            }
        }
    }
}

// ---------------------------------------------------------------------------
// Decoder: x_h -> relu(x_h@W1+b1) @ W2 + b2 -> out[N,6]   (W2 padded 6->16 cols)
// ---------------------------------------------------------------------------
__global__ void k_decode(const float* __restrict__ x_h,
                         const __bf16* __restrict__ W1T, const float* __restrict__ b1,
                         const __bf16* __restrict__ W2T /*[16][64]*/, const float* __restrict__ b2,
                         float* __restrict__ out) {
    __shared__ __bf16 ldsIn[64 * 64];
    __shared__ __bf16 ldsH[64 * 64];
    int tid = threadIdx.x;
    int n0 = blockIdx.x * 64;
    int r = tid >> 1, hf = tid & 1;
    int node = n0 + r;
    #pragma unroll
    for (int j = 0; j < 8; ++j) {
        int c = hf * 32 + j * 4;
        float4 v = make_float4(0.f, 0.f, 0.f, 0.f);
        if (node < N_) v = *reinterpret_cast<const float4*>(x_h + node * 64 + c);
        ldsIn[r * 64 + c + 0] = f2bf(v.x);
        ldsIn[r * 64 + c + 1] = f2bf(v.y);
        ldsIn[r * 64 + c + 2] = f2bf(v.z);
        ldsIn[r * 64 + c + 3] = f2bf(v.w);
    }
    __syncthreads();
    int wv = tid >> 5, lane = tid & 31, m0 = wv * 16;
    wave_mlp_layer1<64>(ldsIn, W1T, b1, ldsH, m0, lane);
    __syncthreads();
    v8f acc = {};
    #pragma unroll
    for (int k0 = 0; k0 < 64; k0 += 32) {
        v16bf a = load_frag_a(ldsH + m0 * 64 + k0, 64, lane);
        v16bf b = load_frag_b(W2T + k0, 64, lane);   // n0 = 0, single 16-col tile
        acc = wmma_bf16(a, b, acc);
    }
    int n = lane & 15, hi = lane >> 4;
    if (n < FOUT_) {
        float bias = b2[n];
        #pragma unroll
        for (int j = 0; j < 8; ++j) {
            int m = hi ? j + 8 : j;
            int nd = n0 + m0 + m;
            if (nd < N_) out[nd * FOUT_ + n] = acc[j] + bias;
        }
    }
}

// ---------------------------------------------------------------------------
extern "C" void kernel_launch(void* const* d_in, const int* in_sizes, int n_in,
                              void* d_out, int out_size, void* d_ws, size_t ws_size,
                              hipStream_t stream) {
    (void)in_sizes; (void)n_in; (void)out_size; (void)ws_size;

    const float* x         = (const float*)d_in[0];
    const float* edge_attr = (const float*)d_in[1];
    const int*   eidx      = (const int*)  d_in[2];
    const float* enc_n_w1 = (const float*)d_in[3];  const float* enc_n_b1 = (const float*)d_in[4];
    const float* enc_n_w2 = (const float*)d_in[5];  const float* enc_n_b2 = (const float*)d_in[6];
    const float* enc_e_w1 = (const float*)d_in[7];  const float* enc_e_b1 = (const float*)d_in[8];
    const float* enc_e_w2 = (const float*)d_in[9];  const float* enc_e_b2 = (const float*)d_in[10];
    const float* proc_e_w1 = (const float*)d_in[11]; const float* proc_e_b1 = (const float*)d_in[12];
    const float* proc_e_w2 = (const float*)d_in[13]; const float* proc_e_b2 = (const float*)d_in[14];
    const float* proc_n_w1 = (const float*)d_in[15]; const float* proc_n_b1 = (const float*)d_in[16];
    const float* proc_n_w2 = (const float*)d_in[17]; const float* proc_n_b2 = (const float*)d_in[18];
    const float* dec_w1 = (const float*)d_in[19]; const float* dec_b1 = (const float*)d_in[20];
    const float* dec_w2 = (const float*)d_in[21]; const float* dec_b2 = (const float*)d_in[22];

    // ---- workspace layout (bytes) ----
    char* ws = (char*)d_ws;
    size_t off = 0;
    float* x_h = (float*)(ws + off); off += (size_t)N_ * 64 * 4;   // 12.8 MB
    float* agg = (float*)(ws + off); off += (size_t)N_ * 64 * 4;   // 12.8 MB
    float* e_h = (float*)(ws + off); off += (size_t)E_ * 64 * 4;   // 204.8 MB
    __bf16* enc_n_w1T = (__bf16*)(ws + off); off += 64 * 32 * 2;
    __bf16* enc_n_w2T = (__bf16*)(ws + off); off += 64 * 64 * 2;
    __bf16* enc_e_w1T = (__bf16*)(ws + off); off += 64 * 32 * 2;
    __bf16* enc_e_w2T = (__bf16*)(ws + off); off += 64 * 64 * 2;
    __bf16* proc_e_w1T = (__bf16*)(ws + off); off += (size_t)L_ * 64 * 192 * 2;
    __bf16* proc_e_w2T = (__bf16*)(ws + off); off += (size_t)L_ * 64 * 64 * 2;
    __bf16* proc_n_w1T = (__bf16*)(ws + off); off += (size_t)L_ * 64 * 128 * 2;
    __bf16* proc_n_w2T = (__bf16*)(ws + off); off += (size_t)L_ * 64 * 64 * 2;
    __bf16* dec_w1T = (__bf16*)(ws + off); off += 64 * 64 * 2;
    __bf16* dec_w2T = (__bf16*)(ws + off); off += 16 * 64 * 2;

    auto cvt = [&](const float* W, __bf16* Wt, int K, int Nc, int Kpad, int Npad) {
        int total = Kpad * Npad;
        k_cvt_wT<<<(total + 255) / 256, 256, 0, stream>>>(W, Wt, K, Nc, Kpad, Npad);
    };
    cvt(enc_n_w1, enc_n_w1T, FN_, 64, 32, 64);
    cvt(enc_n_w2, enc_n_w2T, 64, 64, 64, 64);
    cvt(enc_e_w1, enc_e_w1T, FE_, 64, 32, 64);
    cvt(enc_e_w2, enc_e_w2T, 64, 64, 64, 64);
    for (int l = 0; l < L_; ++l) {
        cvt(proc_e_w1 + (size_t)l * 192 * 64, proc_e_w1T + (size_t)l * 64 * 192, 192, 64, 192, 64);
        cvt(proc_e_w2 + (size_t)l * 64 * 64,  proc_e_w2T + (size_t)l * 64 * 64,  64, 64, 64, 64);
        cvt(proc_n_w1 + (size_t)l * 128 * 64, proc_n_w1T + (size_t)l * 64 * 128, 128, 64, 128, 64);
        cvt(proc_n_w2 + (size_t)l * 64 * 64,  proc_n_w2T + (size_t)l * 64 * 64,  64, 64, 64, 64);
    }
    cvt(dec_w1, dec_w1T, 64, 64, 64, 64);
    cvt(dec_w2, dec_w2T, 64, FOUT_, 64, 16);

    const int nodeBlocks = (N_ + 63) / 64;
    const int edgeBlocks = E_ / 64;

    // encode
    k_encode_nodes<<<nodeBlocks, 128, 0, stream>>>(x, enc_n_w1T, enc_n_b1, enc_n_w2T, enc_n_b2, x_h);
    k_encode_edges<<<edgeBlocks, 128, 0, stream>>>(edge_attr, enc_e_w1T, enc_e_b1, enc_e_w2T, enc_e_b2, e_h);

    // process
    for (int l = 0; l < L_; ++l) {
        k_zero<<<(N_ * 64 + 255) / 256, 256, 0, stream>>>(agg, N_ * 64);
        k_edge_step<<<edgeBlocks, 128, 0, stream>>>(eidx, x_h, e_h, agg,
            proc_e_w1T + (size_t)l * 64 * 192, proc_e_b1 + (size_t)l * 64,
            proc_e_w2T + (size_t)l * 64 * 64,  proc_e_b2 + (size_t)l * 64);
        k_node_step<<<nodeBlocks, 128, 0, stream>>>(agg, x_h,
            proc_n_w1T + (size_t)l * 64 * 128, proc_n_b1 + (size_t)l * 64,
            proc_n_w2T + (size_t)l * 64 * 64,  proc_n_b2 + (size_t)l * 64);
    }

    // decode
    k_decode<<<nodeBlocks, 128, 0, stream>>>(x_h, dec_w1T, dec_b1, dec_w2T, dec_b2, (float*)d_out);
}
